// PlainSpikeSelfAttention_65816078844476
// MI455X (gfx1250) — matrix operand order repaired and checked by reading, not compiled, not run on hardware
//
#include <hip/hip_runtime.h>
#include <stdint.h>

typedef __attribute__((ext_vector_type(16))) __bf16  v16bf;
typedef __attribute__((ext_vector_type(8)))  float   v8f;
typedef __attribute__((ext_vector_type(4)))  unsigned int v4u;

#define T_  4
#define B_  4
#define N_  1024
#define D_  512
#define H_  8
#define HD_ 64
#define M_  (T_ * B_ * N_)   // 16384 rows
#define EPS_ 1e-6f

union ABu { v16bf v; v4u q[2]; unsigned int u[8]; };

__device__ __forceinline__ unsigned short f2bf(float f) {
  unsigned int u = __builtin_bit_cast(unsigned int, f);
  u += 0x7FFFu + ((u >> 16) & 1u);   // round-to-nearest-even
  return (unsigned short)(u >> 16);
}

// 16x32 bf16 A-tile from row-major source (row stride ld, caller pre-offsets to row0/k0).
// ISA layout: lanes 0-15 rows 0-15; VGPR0-3 hold K = 8g..8g+7 pairs, VGPR4-7 hold K = 16+8g..+7.
__device__ __forceinline__ v16bf load_a(const __bf16* src, int ld) {
  int lane = threadIdx.x & 31;
  int g = lane >> 4, r = lane & 15;
  const __bf16* p = src + r * ld + g * 8;
  ABu a;
  a.q[0] = *(const v4u*)(p);        // K = 8g .. 8g+7
  a.q[1] = *(const v4u*)(p + 16);   // K = 16+8g .. 16+8g+7
  return a.v;
}

// 32x16 bf16 B-tile where B(k,n) = src[n*ld + k]  (covers W^T and K^T patterns).
// ISA layout: lane column n = lane&15; VGPR0-3 hold K = 8g..8g+7, VGPR4-7 hold K = 16+8g..+7.
__device__ __forceinline__ v16bf load_bt(const __bf16* src, int ld) {
  int lane = threadIdx.x & 31;
  int g = lane >> 4, n = lane & 15;
  const __bf16* p = src + n * ld + g * 8;
  ABu b;
  b.q[0] = *(const v4u*)(p);
  b.q[1] = *(const v4u*)(p + 16);
  return b.v;
}

// ---------------- pass 0: fp32 -> bf16 ----------------
__global__ __launch_bounds__(256) void cvt_bf16_kernel(const float* __restrict__ src,
                                                       __bf16* __restrict__ dst, int n) {
  int idx = (blockIdx.x * blockDim.x + threadIdx.x) * 4;
  if (idx + 3 < n) {
    float4 f = *(const float4*)(src + idx);
    unsigned int lo = (unsigned int)f2bf(f.x) | ((unsigned int)f2bf(f.y) << 16);
    unsigned int hi = (unsigned int)f2bf(f.z) | ((unsigned int)f2bf(f.w) << 16);
    *(unsigned int*)((unsigned short*)dst + idx)     = lo;
    *(unsigned int*)((unsigned short*)dst + idx + 2) = hi;
  }
}

// ---------------- pass 1: QKV projection, scatter to [tb,h,n,hd] bf16 ----------------
__global__ __launch_bounds__(256) void qkv_kernel(const __bf16* __restrict__ xb,
                                                  const __bf16* __restrict__ wqb,
                                                  const __bf16* __restrict__ wkb,
                                                  const __bf16* __restrict__ wvb,
                                                  const float* __restrict__ bq,
                                                  const float* __restrict__ bk,
                                                  const float* __restrict__ bv,
                                                  __bf16* __restrict__ qh,
                                                  __bf16* __restrict__ kh,
                                                  __bf16* __restrict__ vh) {
  int wave = threadIdx.x >> 5;
  int lane = threadIdx.x & 31;
  int m0 = blockIdx.x * 16;
  int n0 = blockIdx.y * 128 + wave * 16;
  int which = blockIdx.z;
  const __bf16* W   = (which == 0) ? wqb : (which == 1) ? wkb : wvb;
  const float* bias = (which == 0) ? bq  : (which == 1) ? bk  : bv;
  __bf16* out       = (which == 0) ? qh  : (which == 1) ? kh  : vh;

  v8f acc = {0.f, 0.f, 0.f, 0.f, 0.f, 0.f, 0.f, 0.f};
  for (int k = 0; k < D_; k += 32) {
    v16bf a = load_a(xb + (size_t)m0 * D_ + k, D_);
    v16bf b = load_bt(W + (size_t)n0 * D_ + k, D_);
    acc = __builtin_amdgcn_wmma_f32_16x16x32_bf16(false, a, false, b, (short)0, acc, false, false);
  }
  int g = lane >> 4, c = lane & 15;
  int col = n0 + c;
  float bval = bias[col];
  int h = col >> 6, hd = col & 63;
  for (int r = 0; r < 8; ++r) {
    int m  = m0 + g * 8 + r;
    int tb = m >> 10, n = m & (N_ - 1);
    size_t o = ((size_t)(tb * H_ + h) * N_ + n) * HD_ + hd;
    ((unsigned short*)out)[o] = f2bf(acc[r] + bval);
  }
}

// ---------------- pass 2: relu-normalized attention per (t,b,h) ----------------
__global__ __launch_bounds__(128) void attn_kernel(const __bf16* __restrict__ qh,
                                                   const __bf16* __restrict__ kh,
                                                   const __bf16* __restrict__ vh,
                                                   __bf16* __restrict__ ctx) {
  __shared__ __align__(16) __bf16 kt[64 * 72];          // K tile, padded stride
  __shared__ __align__(16) __bf16 vt[64 * 72];          // V tile transposed: vt[hd][key]
  __shared__ __align__(16) __bf16 sst[4 * 16 * 64];     // per-wave S staging

  int tid  = threadIdx.x;
  int wave = tid >> 5, lane = tid & 31;
  int qt   = blockIdx.x;        // 0..15 query tiles of 64
  int tbh  = blockIdx.y;        // 0..127 (t*B+b)*H+h
  int n0   = qt * 64 + wave * 16;

  const __bf16* qbase = qh + ((size_t)tbh * N_ + n0) * HD_;
  v16bf qa0 = load_a(qbase, HD_);        // K(hd) = 0..31
  v16bf qa1 = load_a(qbase + 32, HD_);   // K(hd) = 32..63

  v8f vzero = {0.f, 0.f, 0.f, 0.f, 0.f, 0.f, 0.f, 0.f};
  v8f ctxacc[4] = {vzero, vzero, vzero, vzero};
  v8f rs = vzero;
  __bf16* smy = sst + wave * 16 * 64;
  int g = lane >> 4, c = lane & 15;

  for (int kt0 = 0; kt0 < N_; kt0 += 64) {
    const __bf16* ksrc = kh + ((size_t)tbh * N_ + kt0) * HD_;
    const __bf16* vsrc = vh + ((size_t)tbh * N_ + kt0) * HD_;
    __syncthreads();   // previous iteration's readers of kt/vt are done
    for (int it = 0; it < 4; ++it) {
      int e   = (it * 128 + tid) * 8;
      int row = e >> 6, col = e & 63;                    // key row, hd col
      v4u dk = *(const v4u*)(ksrc + e);
      *(v4u*)(kt + row * 72 + col) = dk;
      v4u dv = *(const v4u*)(vsrc + e);
      union { v4u q; unsigned short s[8]; } u; u.q = dv;
      for (int j = 0; j < 8; ++j)
        ((unsigned short*)vt)[(col + j) * 72 + row] = u.s[j];  // transpose
    }
    __syncthreads();

    // S = relu(Q @ K^T * 1/sqrt(HD)); accumulate rowsums
    v8f s[4];
    for (int j = 0; j < 4; ++j) {
      v8f a = vzero;
      a = __builtin_amdgcn_wmma_f32_16x16x32_bf16(false, qa0, false,
              load_bt(kt + (j * 16) * 72 + 0, 72), (short)0, a, false, false);
      a = __builtin_amdgcn_wmma_f32_16x16x32_bf16(false, qa1, false,
              load_bt(kt + (j * 16) * 72 + 32, 72), (short)0, a, false, false);
      for (int r = 0; r < 8; ++r) {
        float v = a[r] * 0.125f;
        a[r] = v > 0.f ? v : 0.f;
      }
      rs += a;
      s[j] = a;
    }
    // restage S as bf16 in per-wave LDS (row-major 16x64) for A-layout reload
    for (int j = 0; j < 4; ++j)
      for (int r = 0; r < 8; ++r)
        ((unsigned short*)smy)[(g * 8 + r) * 64 + (j * 16 + c)] = f2bf(s[j][r]);

    // ctx += S @ V   (A from smy, B from transposed vt)
    for (int ks = 0; ks < 64; ks += 32) {
      v16bf sa = load_a(smy + ks, 64);
      for (int j = 0; j < 4; ++j) {
        v16bf vb = load_bt(vt + (j * 16) * 72 + ks, 72);
        ctxacc[j] = __builtin_amdgcn_wmma_f32_16x16x32_bf16(false, sa, false, vb,
                                                            (short)0, ctxacc[j], false, false);
      }
    }
  }

  // reduce rowsums across the 16 lanes of each half-wave
  for (int off = 1; off < 16; off <<= 1)
    for (int r = 0; r < 8; ++r)
      rs[r] += __shfl_xor(rs[r], off, 32);

  __bf16* obase = ctx + ((size_t)tbh * N_ + n0) * HD_;
  for (int j = 0; j < 4; ++j)
    for (int r = 0; r < 8; ++r) {
      float v = ctxacc[j][r] / (rs[r] + EPS_);
      ((unsigned short*)obase)[(g * 8 + r) * HD_ + (j * 16 + c)] = f2bf(v);
    }
}

// ---------------- pass 3: output projection, fp32 out ----------------
__global__ __launch_bounds__(256) void outproj_kernel(const __bf16* __restrict__ ctx,
                                                      const __bf16* __restrict__ wob,
                                                      const float* __restrict__ bo,
                                                      float* __restrict__ out) {
  int wave = threadIdx.x >> 5, lane = threadIdx.x & 31;
  int m0 = blockIdx.x * 16;
  int n0 = blockIdx.y * 128 + wave * 16;
  int tb = m0 >> 10, nrow = m0 & (N_ - 1);   // all 16 rows share tb (16 | 1024)

  v8f acc = {0.f, 0.f, 0.f, 0.f, 0.f, 0.f, 0.f, 0.f};
  for (int k0 = 0; k0 < D_; k0 += 32) {
    int h = k0 >> 6;                          // 32-col window stays inside one head
    const __bf16* abase = ctx + (((size_t)(tb * H_ + h)) * N_ + nrow) * HD_ + (k0 & 63);
    v16bf a = load_a(abase, HD_);
    v16bf b = load_bt(wob + (size_t)n0 * D_ + k0, D_);
    acc = __builtin_amdgcn_wmma_f32_16x16x32_bf16(false, a, false, b, (short)0, acc, false, false);
  }
  int g = lane >> 4, c = lane & 15;
  float bias = bo[n0 + c];
  for (int r = 0; r < 8; ++r) {
    int m = m0 + g * 8 + r;
    out[(size_t)m * D_ + n0 + c] = acc[r] + bias;
  }
}

extern "C" void kernel_launch(void* const* d_in, const int* in_sizes, int n_in,
                              void* d_out, int out_size, void* d_ws, size_t ws_size,
                              hipStream_t stream) {
  const float* x  = (const float*)d_in[0];
  const float* wq = (const float*)d_in[1];
  const float* bq = (const float*)d_in[2];
  const float* wk = (const float*)d_in[3];
  const float* bk = (const float*)d_in[4];
  const float* wv = (const float*)d_in[5];
  const float* bv = (const float*)d_in[6];
  const float* wo = (const float*)d_in[7];
  const float* bo = (const float*)d_in[8];

  char* ws = (char*)d_ws;
  const size_t MB = 1u << 20;
  __bf16* xb  = (__bf16*)(ws);               // 16 MB; reused as ctx after pass 1
  __bf16* qh  = (__bf16*)(ws + 16 * MB);
  __bf16* kh  = (__bf16*)(ws + 32 * MB);
  __bf16* vh  = (__bf16*)(ws + 48 * MB);
  __bf16* wqb = (__bf16*)(ws + 64 * MB);
  __bf16* wkb = wqb + (size_t)D_ * D_;
  __bf16* wvb = wkb + (size_t)D_ * D_;
  __bf16* wob = wvb + (size_t)D_ * D_;
  __bf16* ctx = xb;

  const int NX = M_ * D_;        // 8388608
  const int NW = D_ * D_;        // 262144
  cvt_bf16_kernel<<<dim3(NX / 4 / 256), dim3(256), 0, stream>>>(x,  xb,  NX);
  cvt_bf16_kernel<<<dim3(NW / 4 / 256), dim3(256), 0, stream>>>(wq, wqb, NW);
  cvt_bf16_kernel<<<dim3(NW / 4 / 256), dim3(256), 0, stream>>>(wk, wkb, NW);
  cvt_bf16_kernel<<<dim3(NW / 4 / 256), dim3(256), 0, stream>>>(wv, wvb, NW);
  cvt_bf16_kernel<<<dim3(NW / 4 / 256), dim3(256), 0, stream>>>(wo, wob, NW);

  qkv_kernel<<<dim3(M_ / 16, D_ / 128, 3), dim3(256), 0, stream>>>(
      xb, wqb, wkb, wvb, bq, bk, bv, qh, kh, vh);

  attn_kernel<<<dim3(N_ / 64, T_ * B_ * H_), dim3(128), 0, stream>>>(qh, kh, vh, ctx);

  outproj_kernel<<<dim3(M_ / 16, D_ / 128), dim3(256), 0, stream>>>(
      ctx, wob, bo, (float*)d_out);
}
